// GAT_7421703487704
// MI455X (gfx1250) — matrix operand order repaired and checked by reading, not compiled, not run on hardware
//
#include <hip/hip_runtime.h>
#include <hip/hip_bf16.h>
#include <cstdint>

#define DEVINL __device__ __forceinline__

typedef __bf16 bf16;
typedef __attribute__((ext_vector_type(16))) __bf16 v16bf;
typedef __attribute__((ext_vector_type(8)))  float  v8f;

namespace {
constexpr int kB = 8, kN = 1024, kF = 256, kHID = 64, kH = 8, kC = 256;
constexpr int kCAT = kHID * kH;               // 512
constexpr float kALPHA = 0.2f;                // inner leaky slope
constexpr float kOUTER = 0.01f;               // outer leaky slope
constexpr float kL2E   = 1.44269504088896f;   // log2(e)
}

// ---------------------------------------------------------------------------
// WMMA fragment helpers (ISA 7.12.2 layouts, wave32)
// A (16x32 bf16, MxK): lane L holds row M=L&15; K chunks [kb..kb+7] and
//   [16+kb..16+kb+7] with kb = (L>>4)*8  -> two 16B contiguous loads.
// B (32x16 bf16, KxN): lane L holds col N=L&15; K chunks [kb..kb+7] and
//   [16+kb..16+kb+7] with kb = (L>>4)*8 -> needs K-contiguous (transposed) storage.
// C/D (16x16 f32): VGPR r, lane L -> M = r + 8*(L>>4), N = L&15.
// ---------------------------------------------------------------------------
union Frag16 { v16bf v; uint4 q[2]; bf16 e[16]; };

DEVINL v16bf load_a_frag(const bf16* base, int ld, int lane) {
  Frag16 f;
  const bf16* p = base + (size_t)(lane & 15) * ld + ((lane >> 4) << 3);
  f.q[0] = *reinterpret_cast<const uint4*>(p);
  f.q[1] = *reinterpret_cast<const uint4*>(p + 16);
  return f.v;
}

DEVINL v16bf load_bt_frag(const bf16* baseT, int ld, int lane) {
  Frag16 f;
  const bf16* p = baseT + (size_t)(lane & 15) * ld + ((lane >> 4) << 3);
  f.q[0] = *reinterpret_cast<const uint4*>(p);
  f.q[1] = *reinterpret_cast<const uint4*>(p + 16);
  return f.v;
}

DEVINL v8f wmma_bf16(v16bf a, v16bf b, v8f c) {
  return __builtin_amdgcn_wmma_f32_16x16x32_bf16(false, a, false, b,
                                                 (short)0, c, false, false);
}

DEVINL v8f zero8() { v8f z = {0.f,0.f,0.f,0.f,0.f,0.f,0.f,0.f}; return z; }

// ---------------------------------------------------------------------------
// Prep: weight transposes to bf16 (K-contiguous for B fragments), x cast
// ---------------------------------------------------------------------------
__global__ void k_prep_watt(const float* __restrict__ W, bf16* __restrict__ WT) {
  int idx = blockIdx.x * blockDim.x + threadIdx.x;       // H*HID*F
  if (idx >= kH * kHID * kF) return;
  int f = idx % kF; int r = idx / kF; int o = r % kHID; int h = r / kHID;
  WT[((size_t)h * kHID + o) * kF + f] = (bf16)W[((size_t)h * kF + f) * kHID + o];
}

__global__ void k_prep_wout(const float* __restrict__ W, bf16* __restrict__ WT) {
  int idx = blockIdx.x * blockDim.x + threadIdx.x;       // C*CAT
  if (idx >= kC * kCAT) return;
  int f = idx % kCAT; int c = idx / kCAT;
  WT[(size_t)c * kCAT + f] = (bf16)W[(size_t)f * kC + c];
}

__global__ void k_cast(const float* __restrict__ s, bf16* __restrict__ d, int n) {
  int i = blockIdx.x * blockDim.x + threadIdx.x;
  if (i < n) d[i] = (bf16)s[i];
}

// ---------------------------------------------------------------------------
// hT[b][h][o][n] = (x @ W_att)^T  via bf16 WMMA, 1 wave = 16 nodes x 64 hid
// ---------------------------------------------------------------------------
__global__ void __launch_bounds__(32)
k_gemm_h(const bf16* __restrict__ xb, const bf16* __restrict__ WT,
         bf16* __restrict__ hT) {
  int lane = threadIdx.x;
  int blk = blockIdx.x;                          // B*H*(N/16) = 4096
  int nt = blk & 63; int hh = (blk >> 6) & 7; int bi = blk >> 9;
  const bf16* Abase = xb + ((size_t)bi * kN + nt * 16) * kF;
  const bf16* Bbase = WT + (size_t)hh * kHID * kF;
  v8f acc[4]; 
#pragma unroll
  for (int t = 0; t < 4; ++t) acc[t] = zero8();
  for (int k = 0; k < kF; k += 32) {
    v16bf a = load_a_frag(Abase + k, kF, lane);
#pragma unroll
    for (int t = 0; t < 4; ++t) {
      v16bf bb = load_bt_frag(Bbase + (size_t)(t * 16) * kF + k, kF, lane);
      acc[t] = wmma_bf16(a, bb, acc[t]);
    }
  }
  int col = lane & 15, hi = lane >> 4;
  bf16* out = hT + (size_t)(bi * kH + hh) * kHID * kN;
#pragma unroll
  for (int t = 0; t < 4; ++t)
#pragma unroll
    for (int r = 0; r < 8; ++r) {
      int node = nt * 16 + r + 8 * hi;
      out[(size_t)(t * 16 + col) * kN + node] = (bf16)acc[t][r];
    }
}

// f1/f2 per (b,h,n): dot of h row with attention vector halves
__global__ void k_f12(const bf16* __restrict__ hT, const float* __restrict__ a_att,
                      float* __restrict__ f1, float* __restrict__ f2) {
  int idx = blockIdx.x * blockDim.x + threadIdx.x;   // B*H*N
  if (idx >= kB * kH * kN) return;
  int n = idx % kN; int bh = idx / kN; int hh = bh % kH;
  const bf16* base = hT + (size_t)bh * kHID * kN + n;
  const float* a1 = a_att + (size_t)hh * 2 * kHID;
  const float* a2 = a1 + kHID;
  float s1 = 0.f, s2 = 0.f;
  for (int o = 0; o < kHID; ++o) {
    float v = (float)base[(size_t)o * kN];
    s1 += v * a1[o]; s2 += v * a2[o];
  }
  f1[idx] = s1; f2[idx] = s2;
}

// ---------------------------------------------------------------------------
// Fused masked softmax + (att @ h) via WMMA + elu + outer leaky -> y (bf16)
// 1 wave handles a 16-row tile of one (b,h); no attention matrix materialized
// ---------------------------------------------------------------------------
__global__ void __launch_bounds__(32)
k_attn(const bf16* __restrict__ hT, const float* __restrict__ f1,
       const float* __restrict__ f2, const int* __restrict__ adj,
       bf16* __restrict__ y) {
  int lane = threadIdx.x;
  int blk = blockIdx.x;                           // B*H*(N/16)
  int nt = blk & 63; int hh = (blk >> 6) & 7; int bi = blk >> 9;
  int bh = bi * kH + hh;
  const float* F1 = f1 + (size_t)bh * kN;
  const float* F2 = f2 + (size_t)bh * kN;
  const int* A = adj + (size_t)bi * kN * kN + (size_t)nt * 16 * kN;

  // pass 1: per-row max and sum(exp) over masked leaky(f1_i+f2_j)
  float myMax = 0.f, myInv = 0.f;
  for (int m = 0; m < 16; ++m) {
    float f1v = F1[nt * 16 + m];
    const int* Ar = A + (size_t)m * kN;
    float mmax = -3.0e38f;
    for (int j = lane; j < kN; j += 32) {
      float e = f1v + F2[j]; e = e > 0.f ? e : kALPHA * e;
      if (Ar[j] > 0) mmax = fmaxf(mmax, e);
    }
#pragma unroll
    for (int off = 16; off >= 1; off >>= 1) mmax = fmaxf(mmax, __shfl_xor(mmax, off));
    float ssum = 0.f;
    for (int j = lane; j < kN; j += 32) {
      float e = f1v + F2[j]; e = e > 0.f ? e : kALPHA * e;
      if (Ar[j] > 0) ssum += exp2f((e - mmax) * kL2E);
    }
#pragma unroll
    for (int off = 16; off >= 1; off >>= 1) ssum += __shfl_xor(ssum, off);
    if ((lane & 15) == m) { myMax = mmax; myInv = 1.f / fmaxf(ssum, 1e-30f); }
  }

  // pass 2: stream probability tiles into WMMA against hT
  int row = lane & 15;
  int kb8 = (lane >> 4) << 3;
  float f1row = F1[nt * 16 + row];
  const int* Arow = A + (size_t)row * kN;
  const bf16* HT = hT + (size_t)bh * kHID * kN;
  v8f acc[4];
#pragma unroll
  for (int t = 0; t < 4; ++t) acc[t] = zero8();
  for (int jc = 0; jc < kN; jc += 32) {
    Frag16 pf;
#pragma unroll
    for (int hf = 0; hf < 2; ++hf) {
      int k0 = jc + hf * 16 + kb8;
#pragma unroll
      for (int q = 0; q < 8; ++q) {
        int j = k0 + q;
        float e = f1row + F2[j]; e = e > 0.f ? e : kALPHA * e;
        float p = (Arow[j] > 0) ? exp2f((e - myMax) * kL2E) * myInv : 0.f;
        pf.e[hf * 8 + q] = (bf16)p;
      }
    }
#pragma unroll
    for (int t = 0; t < 4; ++t) {
      v16bf bb = load_bt_frag(HT + (size_t)(t * 16) * kN + jc, kN, lane);
      acc[t] = wmma_bf16(pf.v, bb, acc[t]);
    }
  }

  // epilogue: elu, outer leaky(0.01), write concat-head y (bf16)
  int col = lane & 15, hi = lane >> 4;
#pragma unroll
  for (int t = 0; t < 4; ++t)
#pragma unroll
    for (int r = 0; r < 8; ++r) {
      float v = acc[t][r];
      v = v > 0.f ? v : (__expf(v) - 1.f);
      v = v > 0.f ? v : kOUTER * v;
      int node = nt * 16 + r + 8 * hi;
      y[((size_t)bi * kN + node) * kCAT + hh * kHID + t * 16 + col] = (bf16)v;
    }
}

// ---------------------------------------------------------------------------
// ho = y @ W_out (bf16 WMMA). 1 wave: 16 nodes x 64 classes. writes f32 + T bf16
// ---------------------------------------------------------------------------
__global__ void __launch_bounds__(32)
k_gemm_ho(const bf16* __restrict__ y, const bf16* __restrict__ WT,
          float* __restrict__ ho, bf16* __restrict__ hoT) {
  int lane = threadIdx.x;
  int blk = blockIdx.x;                         // B*(N/16)*4 = 2048
  int ct4 = blk & 3; int nt = (blk >> 2) & 63; int bi = blk >> 8;
  const bf16* Abase = y + ((size_t)bi * kN + nt * 16) * kCAT;
  v8f acc[4];
#pragma unroll
  for (int t = 0; t < 4; ++t) acc[t] = zero8();
  for (int k = 0; k < kCAT; k += 32) {
    v16bf a = load_a_frag(Abase + k, kCAT, lane);
#pragma unroll
    for (int t = 0; t < 4; ++t) {
      v16bf bb = load_bt_frag(WT + (size_t)(ct4 * 64 + t * 16) * kCAT + k, kCAT, lane);
      acc[t] = wmma_bf16(a, bb, acc[t]);
    }
  }
  int col = lane & 15, hi = lane >> 4;
#pragma unroll
  for (int t = 0; t < 4; ++t)
#pragma unroll
    for (int r = 0; r < 8; ++r) {
      int node = nt * 16 + r + 8 * hi;
      int c = ct4 * 64 + t * 16 + col;
      float v = acc[t][r];
      ho[((size_t)bi * kN + node) * kC + c] = v;
      hoT[((size_t)bi * kC + c) * kN + node] = (bf16)v;
    }
}

// g1/g2: one wave per (b,n)
__global__ void __launch_bounds__(32)
k_g12(const float* __restrict__ ho, const float* __restrict__ a_out,
      float* __restrict__ g1, float* __restrict__ g2) {
  int bn = blockIdx.x; int lane = threadIdx.x;
  const float* hr = ho + (size_t)bn * kC;
  float s1 = 0.f, s2 = 0.f;
  for (int c = lane; c < kC; c += 32) {
    float v = hr[c]; s1 += v * a_out[c]; s2 += v * a_out[kC + c];
  }
#pragma unroll
  for (int off = 16; off >= 1; off >>= 1) { s1 += __shfl_xor(s1, off); s2 += __shfl_xor(s2, off); }
  if (lane == 0) { g1[bn] = s1; g2[bn] = s2; }
}

// ---------------------------------------------------------------------------
// Fused out-attention: masked softmax + (ao @ ho) via 16-tile WMMA,
// then elu + residual + mask + in-wave LayerNorm -> next x (f32 + bf16)
// ---------------------------------------------------------------------------
__global__ void __launch_bounds__(32)
k_outattn(const bf16* __restrict__ hoT, const float* __restrict__ g1,
          const float* __restrict__ g2, const int* __restrict__ adj,
          const float* __restrict__ xprev, const int* __restrict__ mask,
          const float* __restrict__ lnw, const float* __restrict__ lnb,
          float* __restrict__ xnext, bf16* __restrict__ xbf) {
  int lane = threadIdx.x;
  int blk = blockIdx.x;                          // B*(N/16) = 512
  int nt = blk & 63; int bi = blk >> 6;
  const float* G1 = g1 + (size_t)bi * kN;
  const float* G2 = g2 + (size_t)bi * kN;
  const int* A = adj + (size_t)bi * kN * kN + (size_t)nt * 16 * kN;

  float myMax = 0.f, myInv = 0.f;
  for (int m = 0; m < 16; ++m) {
    float g1v = G1[nt * 16 + m];
    const int* Ar = A + (size_t)m * kN;
    float mmax = -3.0e38f;
    for (int j = lane; j < kN; j += 32) {
      float e = g1v + G2[j]; e = e > 0.f ? e : kALPHA * e;
      if (Ar[j] > 0) mmax = fmaxf(mmax, e);
    }
#pragma unroll
    for (int off = 16; off >= 1; off >>= 1) mmax = fmaxf(mmax, __shfl_xor(mmax, off));
    float ssum = 0.f;
    for (int j = lane; j < kN; j += 32) {
      float e = g1v + G2[j]; e = e > 0.f ? e : kALPHA * e;
      if (Ar[j] > 0) ssum += exp2f((e - mmax) * kL2E);
    }
#pragma unroll
    for (int off = 16; off >= 1; off >>= 1) ssum += __shfl_xor(ssum, off);
    if ((lane & 15) == m) { myMax = mmax; myInv = 1.f / fmaxf(ssum, 1e-30f); }
  }

  int row = lane & 15;
  int kb8 = (lane >> 4) << 3;
  float g1row = G1[nt * 16 + row];
  const int* Arow = A + (size_t)row * kN;
  const bf16* HT = hoT + (size_t)bi * kC * kN;
  v8f acc[16];
#pragma unroll
  for (int t = 0; t < 16; ++t) acc[t] = zero8();
  for (int jc = 0; jc < kN; jc += 32) {
    Frag16 pf;
#pragma unroll
    for (int hf = 0; hf < 2; ++hf) {
      int k0 = jc + hf * 16 + kb8;
#pragma unroll
      for (int q = 0; q < 8; ++q) {
        int j = k0 + q;
        float e = g1row + G2[j]; e = e > 0.f ? e : kALPHA * e;
        float p = (Arow[j] > 0) ? exp2f((e - myMax) * kL2E) * myInv : 0.f;
        pf.e[hf * 8 + q] = (bf16)p;
      }
    }
#pragma unroll
    for (int t = 0; t < 16; ++t) {
      v16bf bb = load_bt_frag(HT + (size_t)(t * 16) * kN + jc, kN, lane);
      acc[t] = wmma_bf16(pf.v, bb, acc[t]);
    }
  }

  // epilogue: t = mask ? xprev + elu(acc) : 0; then LayerNorm over 256 cols
  int col = lane & 15, hi = lane >> 4;
  float srow[8] = {0,0,0,0,0,0,0,0};
  float sq[8]   = {0,0,0,0,0,0,0,0};
#pragma unroll
  for (int t = 0; t < 16; ++t)
#pragma unroll
    for (int r = 0; r < 8; ++r) {
      int node = nt * 16 + r + 8 * hi;
      float v = acc[t][r];
      v = v > 0.f ? v : (__expf(v) - 1.f);
      float tt = xprev[((size_t)bi * kN + node) * kC + t * 16 + col] + v;
      if (mask[bi * kN + node] == 0) tt = 0.f;
      acc[t][r] = tt;
      srow[r] += tt; sq[r] += tt * tt;
    }
  // cross-lane reduce over the 16 cols held in each 16-lane half
#pragma unroll
  for (int r = 0; r < 8; ++r) {
#pragma unroll
    for (int off = 1; off <= 8; off <<= 1) {
      srow[r] += __shfl_xor(srow[r], off);
      sq[r]   += __shfl_xor(sq[r], off);
    }
  }
  float mu[8], rstd[8];
#pragma unroll
  for (int r = 0; r < 8; ++r) {
    mu[r] = srow[r] * (1.f / kC);
    float var = sq[r] * (1.f / kC) - mu[r] * mu[r];
    rstd[r] = rsqrtf(var + 1e-5f);
  }
#pragma unroll
  for (int t = 0; t < 16; ++t)
#pragma unroll
    for (int r = 0; r < 8; ++r) {
      int node = nt * 16 + r + 8 * hi;
      int c = t * 16 + col;
      float o = (acc[t][r] - mu[r]) * rstd[r] * lnw[c] + lnb[c];
      size_t idx = ((size_t)bi * kN + node) * kC + c;
      xnext[idx] = o;
      xbf[idx] = (bf16)o;
    }
}

// final outputs: relu(x) and relu(x[:,0,:]) concatenated flat
__global__ void k_final(const float* __restrict__ xf, float* __restrict__ out) {
  int idx = blockIdx.x * blockDim.x + threadIdx.x;
  const int total = kB * kN * kC;
  if (idx < total) {
    out[idx] = fmaxf(xf[idx], 0.f);
  } else if (idx < total + kB * kC) {
    int r = idx - total; int bi = r / kC; int c = r % kC;
    out[idx] = fmaxf(xf[((size_t)bi * kN) * kC + c], 0.f);
  }
}

// ---------------------------------------------------------------------------
extern "C" void kernel_launch(void* const* d_in, const int* in_sizes, int n_in,
                              void* d_out, int out_size, void* d_ws, size_t ws_size,
                              hipStream_t stream) {
  const float* x_in  = (const float*)d_in[0];
  const int*   adj   = (const int*)d_in[1];
  const int*   mask  = (const int*)d_in[2];
  const float* W_att = (const float*)d_in[3];
  const float* a_att = (const float*)d_in[4];
  const float* W_out = (const float*)d_in[5];
  const float* a_out = (const float*)d_in[6];
  const float* ln_w  = (const float*)d_in[7];
  const float* ln_b  = (const float*)d_in[8];

  char* w = (char*)d_ws; size_t off = 0;
  auto alloc = [&](size_t bytes) -> void* {
    void* p = w + off; off += (bytes + 255) & ~(size_t)255; return p;
  };
  bf16*  WT_att = (bf16*)alloc((size_t)kH * kHID * kF * 2);
  bf16*  WT_out = (bf16*)alloc((size_t)kC * kCAT * 2);
  bf16*  xbf    = (bf16*)alloc((size_t)kB * kN * kF * 2);
  bf16*  hT     = (bf16*)alloc((size_t)kB * kH * kHID * kN * 2);
  float* f1     = (float*)alloc((size_t)kB * kH * kN * 4);
  float* f2     = (float*)alloc((size_t)kB * kH * kN * 4);
  bf16*  yb     = (bf16*)alloc((size_t)kB * kN * kCAT * 2);
  float* ho     = (float*)alloc((size_t)kB * kN * kC * 4);
  bf16*  hoT    = (bf16*)alloc((size_t)kB * kC * kN * 2);
  float* g1     = (float*)alloc((size_t)kB * kN * 4);
  float* g2     = (float*)alloc((size_t)kB * kN * 4);
  float* xa     = (float*)alloc((size_t)kB * kN * kC * 4);
  float* xb2    = (float*)alloc((size_t)kB * kN * kC * 4);

  k_prep_watt<<<(kH * kHID * kF + 255) / 256, 256, 0, stream>>>(W_att, WT_att);
  k_prep_wout<<<(kC * kCAT + 255) / 256, 256, 0, stream>>>(W_out, WT_out);
  k_cast<<<(kB * kN * kF + 255) / 256, 256, 0, stream>>>(x_in, xbf, kB * kN * kF);

  const float* xcur = x_in;
  float* xn[2] = {xa, xb2};
  for (int layer = 0; layer < 2; ++layer) {
    k_gemm_h<<<kB * kH * (kN / 16), 32, 0, stream>>>(xbf, WT_att, hT);
    k_f12<<<(kB * kH * kN + 255) / 256, 256, 0, stream>>>(hT, a_att, f1, f2);
    k_attn<<<kB * kH * (kN / 16), 32, 0, stream>>>(hT, f1, f2, adj, yb);
    k_gemm_ho<<<kB * (kN / 16) * 4, 32, 0, stream>>>(yb, WT_out, ho, hoT);
    k_g12<<<kB * kN, 32, 0, stream>>>(ho, a_out, g1, g2);
    k_outattn<<<kB * (kN / 16), 32, 0, stream>>>(hoT, g1, g2, adj, xcur, mask,
                                                 ln_w, ln_b, xn[layer], xbf);
    xcur = xn[layer];
  }
  k_final<<<(kB * kN * kC + kB * kC + 255) / 256, 256, 0, stream>>>(xcur, (float*)d_out);
}
